// Net_int_13580686590258
// MI455X (gfx1250) — matrix operand order repaired and checked by reading, not compiled, not run on hardware
//
#include <hip/hip_runtime.h>
#include <hip/hip_bf16.h>

// ---------------------------------------------------------------------------
// Problem constants (from reference)
// ---------------------------------------------------------------------------
#define NN   4000      // nodes
#define NE   64000     // edges
#define NP   200000    // readout pairs
#define DIM  64
#define HID  128       // edge-MLP hidden
#define KO   8192      // 128*64 (k,o) flattened
#define LDM  8256      // KO + DIM bias columns

typedef __bf16 bf16;
typedef __attribute__((ext_vector_type(16))) __bf16 v16bf;
typedef __attribute__((ext_vector_type(8)))  __bf16 v8bf;
typedef __attribute__((ext_vector_type(2)))  __bf16 v2bf;
typedef __attribute__((ext_vector_type(8)))  float  v8f;

__device__ __forceinline__ float sigmoidf_(float x) { return 1.0f / (1.0f + __expf(-x)); }

// ---------------------------------------------------------------------------
// K1: out = relu(x @ Wn + bn_)   -> h (f32) and hbf (bf16, WMMA A operand)
// ---------------------------------------------------------------------------
__global__ __launch_bounds__(256) void k_node_embed(
    const float* __restrict__ x, const float* __restrict__ Wn,
    const float* __restrict__ bn_, float* __restrict__ h, bf16* __restrict__ hbf)
{
    int idx = blockIdx.x * blockDim.x + threadIdx.x;
    if (idx >= NN * DIM) return;
    int n = idx >> 6, d = idx & 63;
    float acc = bn_[d];
    #pragma unroll
    for (int q = 0; q < 8; ++q) acc += x[n * 8 + q] * Wn[q * DIM + d];
    acc = fmaxf(acc, 0.0f);
    h[idx]   = acc;
    hbf[idx] = (bf16)acc;
}

// ---------------------------------------------------------------------------
// K2: ea = relu(edge_attr@We+be); h1 = relu(ea@W1+b1)  -> h1 bf16 [NE,128]
// one wave per edge, 8 edges / 256-thread block
// ---------------------------------------------------------------------------
__global__ __launch_bounds__(256) void k_edge_h1(
    const float* __restrict__ edge_attr, const float* __restrict__ We,
    const float* __restrict__ be, const float* __restrict__ W1,
    const float* __restrict__ b1, bf16* __restrict__ h1)
{
    __shared__ float sEA[8][12];
    int wave = threadIdx.x >> 5, lane = threadIdx.x & 31;
    int e = blockIdx.x * 8 + wave;
    if (lane < 12) {
        float acc = be[lane];
        #pragma unroll
        for (int q = 0; q < 19; ++q) acc += edge_attr[e * 19 + q] * We[q * 12 + lane];
        sEA[wave][lane] = fmaxf(acc, 0.0f);
    }
    __syncthreads();
    #pragma unroll
    for (int t = 0; t < 4; ++t) {
        int j = lane + 32 * t;
        float acc = b1[j];
        #pragma unroll
        for (int q = 0; q < 12; ++q) acc += sEA[wave][q] * W1[q * HID + j];
        h1[(size_t)e * HID + j] = (bf16)fmaxf(acc, 0.0f);
    }
}

// ---------------------------------------------------------------------------
// K3: build Bt[ko][i] (bf16, K-contiguous "B transposed" for the WMMA GEMM):
//   ko = k*64+o < 8192 : Bt = W2[k, i*64+o]
//   ko = 8192+o        : Bt = b2[i*64+o]      (bias columns)
// ---------------------------------------------------------------------------
__global__ __launch_bounds__(256) void k_w2_permute(
    const float* __restrict__ W2, const float* __restrict__ b2, bf16* __restrict__ Bt)
{
    int idx = blockIdx.x * blockDim.x + threadIdx.x;
    if (idx >= LDM * DIM) return;
    int ko = idx >> 6, i = idx & 63;
    float v;
    if (ko < KO) {
        int k = ko >> 6, o = ko & 63;
        v = W2[(size_t)k * 4096 + i * 64 + o];
    } else {
        int o = ko - KO;
        v = b2[i * 64 + o];
    }
    Bt[(size_t)ko * 64 + i] = (bf16)v;
}

// ---------------------------------------------------------------------------
// generic zero + degree accumulation
// ---------------------------------------------------------------------------
__global__ void k_zero(float* __restrict__ p, int n)
{
    int i = blockIdx.x * blockDim.x + threadIdx.x;
    if (i < n) p[i] = 0.0f;
}

__global__ void k_deg(const int* __restrict__ ei, float* __restrict__ deg)
{
    int e = blockIdx.x * blockDim.x + threadIdx.x;
    if (e < NE) atomicAdd(&deg[ei[NE + e]], 1.0f);
}

// ---------------------------------------------------------------------------
// K4a: M = hbf @ Bt'   (GEMM  [4000,64] x [64,8256], K=64)
// v_wmma_f32_16x16x32_bf16, one wave per 16x64 output tile, bf16 output.
// A frag per lane: contiguous 16-elem K run at kbase+16g.
// B frag per lane: contiguous 16-elem K run at kbase+16g (Bt is K-contiguous).
// ---------------------------------------------------------------------------
__global__ __launch_bounds__(256) void k_gemm_M(
    const bf16* __restrict__ A, const bf16* __restrict__ Bt, bf16* __restrict__ Mo)
{
    const int lane = threadIdx.x & 31;
    const int wid  = (blockIdx.x * blockDim.x + threadIdx.x) >> 5;
    if (wid >= 250 * 129) return;                 // wave-uniform guard, EXEC stays all-1s
    const int rowTile = wid / 129;
    const int colQuad = wid - rowTile * 129;
    const int row0 = rowTile * 16;
    const int col0 = colQuad * 64;
    const int g = lane >> 4;
    const int m = row0 + (lane & 15);
    const int n = lane & 15;

    const bf16* arow = A + (size_t)m * DIM;
    v8bf a0lo = *(const v8bf*)(arow + 0  + 16 * g);
    v8bf a0hi = *(const v8bf*)(arow + 8  + 16 * g);
    v8bf a1lo = *(const v8bf*)(arow + 32 + 16 * g);
    v8bf a1hi = *(const v8bf*)(arow + 40 + 16 * g);
    v16bf a0, a1;
    #pragma unroll
    for (int i = 0; i < 8; ++i) {
        a0[i] = a0lo[i]; a0[i + 8] = a0hi[i];
        a1[i] = a1lo[i]; a1[i + 8] = a1hi[i];
    }

    #pragma unroll
    for (int ct = 0; ct < 4; ++ct) {
        const int c0 = col0 + ct * 16;
        const bf16* brow = Bt + (size_t)(c0 + n) * DIM + 16 * g;
        v8bf b0a = *(const v8bf*)(brow + 0);
        v8bf b0b = *(const v8bf*)(brow + 8);
        v8bf b1a = *(const v8bf*)(brow + 32);
        v8bf b1b = *(const v8bf*)(brow + 40);
        v16bf b0, b1;
        #pragma unroll
        for (int i = 0; i < 8; ++i) {
            b0[i] = b0a[i]; b0[i + 8] = b0b[i];
            b1[i] = b1a[i]; b1[i + 8] = b1b[i];
        }
        v8f c = {0.f, 0.f, 0.f, 0.f, 0.f, 0.f, 0.f, 0.f};
        c = __builtin_amdgcn_wmma_f32_16x16x32_bf16(false, a0, false, b0, (short)0, c, false, false);
        c = __builtin_amdgcn_wmma_f32_16x16x32_bf16(false, a1, false, b1, (short)0, c, false, false);
        const int col = c0 + n;
        #pragma unroll
        for (int r = 0; r < 8; ++r) {
            int rr = row0 + r + 8 * g;
            Mo[(size_t)rr * LDM + col] = (bf16)c[r];
        }
    }
}

// ---------------------------------------------------------------------------
// K4b: msg[e,:] = h1[e,:] @ M[src[e]]  (+ bias cols), scatter-add into agg[dst]
// one wave per edge; M block is 16 KB contiguous and L2-resident (65 MB << 192 MB L2)
// ---------------------------------------------------------------------------
__global__ __launch_bounds__(256) void k_edge_msg(
    const bf16* __restrict__ h1, const bf16* __restrict__ M,
    const int* __restrict__ ei, float* __restrict__ agg)
{
    __shared__ float sh1[8][HID];
    int wave = threadIdx.x >> 5, lane = threadIdx.x & 31;
    int e = blockIdx.x * 8 + wave;
    #pragma unroll
    for (int t = 0; t < 4; ++t)
        sh1[wave][lane + 32 * t] = (float)h1[(size_t)e * HID + lane + 32 * t];
    __syncthreads();

    int src = ei[e];
    int dst = ei[NE + e];
    const bf16* Mp = M + (size_t)src * LDM;
    __builtin_prefetch((const void*)(Mp + lane * 512), 0, 1);   // pull the 16 KB block

    const int o = 2 * lane;
    float acc0 = 0.0f, acc1 = 0.0f;
    #pragma unroll 8
    for (int k = 0; k < HID; ++k) {
        v2bf mp = *(const v2bf*)(Mp + k * DIM + o);
        float hk = sh1[wave][k];
        acc0 += hk * (float)mp[0];
        acc1 += hk * (float)mp[1];
    }
    v2bf bp = *(const v2bf*)(Mp + KO + o);        // per-node bias term (b2 contraction)
    acc0 += (float)bp[0];
    acc1 += (float)bp[1];
    atomicAdd(&agg[(size_t)dst * DIM + o],     acc0);
    atomicAdd(&agg[(size_t)dst * DIM + o + 1], acc1);
}

// ---------------------------------------------------------------------------
// K4c: per-node GRU update. one wave per node, 8 nodes / block.
// ---------------------------------------------------------------------------
__global__ __launch_bounds__(256) void k_gru(
    float* __restrict__ h, bf16* __restrict__ hbf,
    const float* __restrict__ agg, const float* __restrict__ deg,
    const float* __restrict__ conv_b,
    const float* __restrict__ W_ih, const float* __restrict__ b_ih,
    const float* __restrict__ W_hh, const float* __restrict__ b_hh)
{
    __shared__ float sm[8][DIM], sh[8][DIM], sgi[8][3 * DIM], sgh[8][3 * DIM];
    int wave = threadIdx.x >> 5, lane = threadIdx.x & 31;
    int n = blockIdx.x * 8 + wave;
    float rdg = 1.0f / fmaxf(deg[n], 1.0f);
    #pragma unroll
    for (int t = 0; t < 2; ++t) {
        int d = lane + 32 * t;
        float a = agg[(size_t)n * DIM + d] * rdg + conv_b[d];
        sm[wave][d] = fmaxf(a, 0.0f);
        sh[wave][d] = h[(size_t)n * DIM + d];
    }
    __syncthreads();
    #pragma unroll
    for (int t = 0; t < 6; ++t) {
        int j = lane + 32 * t;
        float gi = b_ih[j], gh = b_hh[j];
        const float4* wi4 = (const float4*)(W_ih + (size_t)j * DIM);
        const float4* wh4 = (const float4*)(W_hh + (size_t)j * DIM);
        #pragma unroll
        for (int d4 = 0; d4 < 16; ++d4) {
            float4 wi = wi4[d4], wh = wh4[d4];
            int d = d4 * 4;
            gi += sm[wave][d] * wi.x + sm[wave][d + 1] * wi.y
                + sm[wave][d + 2] * wi.z + sm[wave][d + 3] * wi.w;
            gh += sh[wave][d] * wh.x + sh[wave][d + 1] * wh.y
                + sh[wave][d + 2] * wh.z + sh[wave][d + 3] * wh.w;
        }
        sgi[wave][j] = gi;
        sgh[wave][j] = gh;
    }
    __syncthreads();
    #pragma unroll
    for (int t = 0; t < 2; ++t) {
        int d = lane + 32 * t;
        float r  = sigmoidf_(sgi[wave][d]       + sgh[wave][d]);
        float z  = sigmoidf_(sgi[wave][64 + d]  + sgh[wave][64 + d]);
        float nn = tanhf(sgi[wave][128 + d] + r * sgh[wave][128 + d]);
        float hn = (1.0f - z) * nn + z * sh[wave][d];
        h[(size_t)n * DIM + d]   = hn;
        hbf[(size_t)n * DIM + d] = (bf16)hn;
    }
}

// ---------------------------------------------------------------------------
// K5: fused readout. one wave per pair p; wave32 shfl_xor reduction.
// ---------------------------------------------------------------------------
__global__ __launch_bounds__(256) void k_readout(
    const float* __restrict__ h, const int* __restrict__ ei3,
    const float* __restrict__ ea3,
    const float* __restrict__ bn_mean, const float* __restrict__ bn_var,
    const float* __restrict__ bn_gamma, const float* __restrict__ bn_beta,
    const float* __restrict__ Wwt, const float* __restrict__ Wbias,
    float* __restrict__ y)
{
    int wave = threadIdx.x >> 5, lane = threadIdx.x & 31;
    int p = blockIdx.x * 8 + wave;
    int i0 = ei3[p], i1 = ei3[NP + p];
    float t0a = h[(size_t)i0 * DIM + lane], t0b = h[(size_t)i0 * DIM + lane + 32];
    float t1a = h[(size_t)i1 * DIM + lane], t1b = h[(size_t)i1 * DIM + lane + 32];
    float eav[8];
    #pragma unroll
    for (int q = 0; q < 8; ++q) eav[q] = ea3[(size_t)p * 8 + q];

    float acc = 0.0f;
    #pragma unroll
    for (int t = 0; t < 6; ++t) {
        int j = lane + 32 * t;
        float t0 = (t & 1) ? t0b : t0a;
        float t1 = (t & 1) ? t1b : t1a;
        float f;
        if (t < 2)      f = 0.5f * (t0 + t1);
        else if (t < 4) f = t0 * t1;
        else { float dd = t0 - t1; f = dd * dd; }
        f = (f - bn_mean[j]) * rsqrtf(bn_var[j] + 1e-5f) * bn_gamma[j] + bn_beta[j];
        float w = 0.0f;
        #pragma unroll
        for (int q = 0; q < 8; ++q) w += eav[q] * Wwt[q * 192 + j];
        acc += f * w;
    }
    #pragma unroll
    for (int off = 16; off > 0; off >>= 1) acc += __shfl_xor(acc, off, 32);
    if (lane == 0) {
        float b = 0.0f;
        #pragma unroll
        for (int q = 0; q < 8; ++q) b += eav[q] * Wbias[q];
        y[p] = acc + b;
    }
}

// ---------------------------------------------------------------------------
// launch
// ---------------------------------------------------------------------------
extern "C" void kernel_launch(void* const* d_in, const int* in_sizes, int n_in,
                              void* d_out, int out_size, void* d_ws, size_t ws_size,
                              hipStream_t stream)
{
    (void)in_sizes; (void)n_in; (void)out_size; (void)ws_size;
    const float* x         = (const float*)d_in[0];
    const float* edge_attr = (const float*)d_in[1];
    const float* ea3       = (const float*)d_in[2];
    const int*   ei        = (const int*)  d_in[3];
    const int*   ei3       = (const int*)  d_in[4];
    const float* Wn        = (const float*)d_in[5];
    const float* bn_       = (const float*)d_in[6];
    const float* We        = (const float*)d_in[7];
    const float* be        = (const float*)d_in[8];
    const float* W1        = (const float*)d_in[9];
    const float* b1        = (const float*)d_in[10];
    const float* W2        = (const float*)d_in[11];
    const float* b2        = (const float*)d_in[12];
    const float* conv_b    = (const float*)d_in[13];
    const float* W_ih      = (const float*)d_in[14];
    const float* b_ih      = (const float*)d_in[15];
    const float* W_hh      = (const float*)d_in[16];
    const float* b_hh      = (const float*)d_in[17];
    const float* bn_mean   = (const float*)d_in[18];
    const float* bn_var    = (const float*)d_in[19];
    const float* bn_gamma  = (const float*)d_in[20];
    const float* bn_beta   = (const float*)d_in[21];
    const float* Wwt       = (const float*)d_in[22];
    const float* Wbias     = (const float*)d_in[23];
    float* yhat = (float*)d_out;

    char* ws = (char*)d_ws;
    float* h    = (float*)(ws);                       // [4000,64] f32     1,024,000 B
    bf16*  hbf  = (bf16*) (ws + 1024000);             // [4000,64] bf16      512,000 B
    bf16*  h1   = (bf16*) (ws + 1536000);             // [64000,128] bf16 16,384,000 B
    bf16*  Bt   = (bf16*) (ws + 17920000);            // [8256,64] bf16    1,056,768 B
    bf16*  Mmat = (bf16*) (ws + 18976768);            // [4000,8256] bf16 66,048,000 B
    float* agg  = (float*)(ws + 85024768);            // [4000,64] f32     1,024,000 B
    float* deg  = (float*)(ws + 86048768);            // [4000] f32           16,000 B

    k_node_embed<<<(NN * DIM + 255) / 256, 256, 0, stream>>>(x, Wn, bn_, h, hbf);
    k_edge_h1<<<NE / 8, 256, 0, stream>>>(edge_attr, We, be, W1, b1, h1);
    k_w2_permute<<<(LDM * DIM + 255) / 256, 256, 0, stream>>>(W2, b2, Bt);
    k_zero<<<(NN + 255) / 256, 256, 0, stream>>>(deg, NN);
    k_deg<<<(NE + 255) / 256, 256, 0, stream>>>(ei, deg);

    for (int it = 0; it < 3; ++it) {
        k_gemm_M<<<(250 * 129 + 7) / 8, 256, 0, stream>>>(hbf, Bt, Mmat);
        k_zero<<<(NN * DIM + 255) / 256, 256, 0, stream>>>(agg, NN * DIM);
        k_edge_msg<<<NE / 8, 256, 0, stream>>>(h1, Mmat, ei, agg);
        k_gru<<<NN / 8, 256, 0, stream>>>(h, hbf, agg, deg, conv_b,
                                          W_ih, b_ih, W_hh, b_hh);
    }

    k_readout<<<NP / 8, 256, 0, stream>>>(h, ei3, ea3, bn_mean, bn_var,
                                          bn_gamma, bn_beta, Wwt, Wbias, yhat);
}
